// Category_guided_Feature_Generation_65197603553364
// MI455X (gfx1250) — matrix operand order repaired and checked by reading, not compiled, not run on hardware
//
#include <hip/hip_runtime.h>
#include <hip/hip_bf16.h>

typedef __attribute__((ext_vector_type(16))) _Float16 v16h;
typedef __attribute__((ext_vector_type(8)))  _Float16 v8h;
typedef __attribute__((ext_vector_type(8)))  float    v8f;
typedef __attribute__((ext_vector_type(4)))  float    v4f;

#define BATCH 16
#define CIN   256
#define NPIX  16384   // 128*128
#define COUT  64
#define KPROT 5

static __device__ __forceinline__ v8f wmma_f16(v16h a, v16h b, v8f c) {
  return __builtin_amdgcn_wmma_f32_16x16x32_f16(false, a, false, b, (short)0, c, false, false);
}
static __device__ __forceinline__ v16h frag_split(const _Float16* p0, const _Float16* p1) {
  v8h lo = *(const v8h*)p0;
  v8h hi = *(const v8h*)p1;
  return __builtin_shufflevector(lo, hi, 0,1,2,3,4,5,6,7,8,9,10,11,12,13,14,15);
}
static __device__ __forceinline__ v8f zero8() {
  v8f z = {0.f,0.f,0.f,0.f,0.f,0.f,0.f,0.f};
  return z;
}

// ---------------------------------------------------------------------------
// Setup: convert all conv weights to f16, fold conv-bias + BN into per-channel
// scale/offset:  y = relu(dot*s + o),  s = g*rsqrt(v+eps), o = be + (bias-m)*s
// wh layout (halfs): W0a[0,16384) W0b[16384,20480) W1a[20480,36864)
//                    W1b[36864,40960) W2[40960,49152)
// so layout (f32):   [layer][ s:0..63 | o:64..127 ],  layers 0..4
// ---------------------------------------------------------------------------
__global__ void __launch_bounds__(256) k_setup(
    const float* __restrict__ W0a, const float* __restrict__ b0a, const float* __restrict__ bn0a,
    const float* __restrict__ W0b, const float* __restrict__ b0b, const float* __restrict__ bn0b,
    const float* __restrict__ W1a, const float* __restrict__ b1a, const float* __restrict__ bn1a,
    const float* __restrict__ W1b, const float* __restrict__ b1b, const float* __restrict__ bn1b,
    const float* __restrict__ W2,  const float* __restrict__ b2,  const float* __restrict__ bn2,
    _Float16* __restrict__ wh, float* __restrict__ so) {
  int t = threadIdx.x;
  for (int i = t; i < 16384; i += 256) wh[i]         = (_Float16)W0a[i];
  for (int i = t; i < 4096;  i += 256) wh[16384 + i] = (_Float16)W0b[i];
  for (int i = t; i < 16384; i += 256) wh[20480 + i] = (_Float16)W1a[i];
  for (int i = t; i < 4096;  i += 256) wh[36864 + i] = (_Float16)W1b[i];
  for (int i = t; i < 8192;  i += 256) wh[40960 + i] = (_Float16)W2[i];
  if (t < 64) {
    int c = t;
    auto fold = [&](int L, const float* bias, const float* bn) {
      float g = bn[c], be = bn[64 + c], m = bn[128 + c], v = bn[192 + c];
      float s = g * rsqrtf(v + 1e-5f);
      so[L * 128 + c]      = s;
      so[L * 128 + 64 + c] = be + (bias[c] - m) * s;
    };
    fold(0, b0a, bn0a); fold(1, b0b, bn0b);
    fold(2, b1a, bn1a); fold(3, b1b, bn1b);
    fold(4, b2, bn2);
  }
}

// ---------------------------------------------------------------------------
// Fused dual-stem: reads x once, computes feats0 = stem0(x), feats1 = stem1(x)
// feats layout: [B][N][64] f16.  Block = 128 pixels, 256 threads = 8 waves,
// each wave owns 16 rows x 64 cols (4 WMMA n-tiles).
// ---------------------------------------------------------------------------
#define XS_STRIDE 40   // halfs, padded row stride for 32-wide x stage
#define HS_STRIDE 72   // halfs, padded row stride for 64-wide h stage
__global__ void __launch_bounds__(256) k_stem(
    const float* __restrict__ x,          // [B][256][N]
    const _Float16* __restrict__ Wa0, const _Float16* __restrict__ Wb0,
    const _Float16* __restrict__ Wa1, const _Float16* __restrict__ Wb1,
    const float* __restrict__ so,
    _Float16* __restrict__ f0, _Float16* __restrict__ f1) {
  __shared__ __align__(16) _Float16 lds[128 * HS_STRIDE];
  const int t = threadIdx.x;
  const int wave = t >> 5, lane = t & 31;
  const int ln = lane & 15, hf = lane >> 4;
  const long pixbase = (long)blockIdx.x * 128;      // flat pixel index b*N+n
  const int b  = (int)(pixbase >> 14);
  const int n0 = (int)(pixbase & (NPIX - 1));
  const float* xb = x + (long)b * CIN * NPIX + n0;

  v8f acc0[4], acc1[4];
#pragma unroll
  for (int j = 0; j < 4; ++j) { acc0[j] = zero8(); acc1[j] = zero8(); }

  // ---- first 1x1 conv for both branches, K = 256 in 8 chunks of 32 ----
  for (int kc = 0; kc < 8; ++kc) {
    __syncthreads();
    // stage x chunk transposed into LDS: xs[pixel][k]  (global reads coalesced)
#pragma unroll
    for (int i = 0; i < 16; ++i) {
      int e = t + i * 256;                 // 0..4095
      int c = e >> 7, p = e & 127;
      float v = xb[(long)(kc * 32 + c) * NPIX + p];
      lds[p * XS_STRIDE + c] = (_Float16)v;
    }
    __syncthreads();
    const int m = wave * 16 + ln;
    const _Float16* ap = &lds[m * XS_STRIDE + hf * 8];
    v16h a = frag_split(ap, ap + 16);
#pragma unroll
    for (int j = 0; j < 4; ++j) {
      int nn = j * 16 + ln;
      v16h bb0 = frag_split(Wa0 + (long)nn * CIN + kc * 32 + hf * 8, Wa0 + (long)nn * CIN + kc * 32 + hf * 8 + 16);
      acc0[j] = wmma_f16(a, bb0, acc0[j]);
      v16h bb1 = frag_split(Wa1 + (long)nn * CIN + kc * 32 + hf * 8, Wa1 + (long)nn * CIN + kc * 32 + hf * 8 + 16);
      acc1[j] = wmma_f16(a, bb1, acc1[j]);
    }
  }

  // ---- each branch: BN+ReLU -> LDS (f16) -> second GEMM -> BN+ReLU -> feats0/feats1 --
#pragma unroll
  for (int br = 0; br < 2; ++br) {
    const float* soa      = so + (br ? 256 : 0);       // layer 0 or 2
    const float* sob      = so + (br ? 384 : 128);     // layer 1 or 3
    const _Float16* Wb    = br ? Wb1 : Wb0;
    _Float16* fo          = br ? f1  : f0;
    __syncthreads();
#pragma unroll
    for (int j = 0; j < 4; ++j) {
      int c = j * 16 + ln;
      float s = soa[c], o = soa[64 + c];
      v8f av = br ? acc1[j] : acc0[j];
#pragma unroll
      for (int r = 0; r < 8; ++r) {
        float h = av[r] * s + o;
        h = h > 0.f ? h : 0.f;
        lds[(wave * 16 + hf * 8 + r) * HS_STRIDE + c] = (_Float16)h;
      }
    }
    __syncthreads();
    v8f d[4];
#pragma unroll
    for (int j = 0; j < 4; ++j) d[j] = zero8();
#pragma unroll
    for (int kc = 0; kc < 2; ++kc) {
      const _Float16* ap = &lds[(wave * 16 + ln) * HS_STRIDE + kc * 32 + hf * 8];
      v16h a = frag_split(ap, ap + 16);
#pragma unroll
      for (int j = 0; j < 4; ++j) {
        v16h bb = frag_split(Wb + (long)(j * 16 + ln) * 64 + kc * 32 + hf * 8, Wb + (long)(j * 16 + ln) * 64 + kc * 32 + hf * 8 + 16);
        d[j] = wmma_f16(a, bb, d[j]);
      }
    }
    _Float16* outp = fo + pixbase * COUT;
#pragma unroll
    for (int j = 0; j < 4; ++j) {
      int c = j * 16 + ln;
      float s = sob[c], o = sob[64 + c];
#pragma unroll
      for (int r = 0; r < 8; ++r) {
        float h = d[j][r] * s + o;
        h = h > 0.f ? h : 0.f;
        outp[(long)(wave * 16 + hf * 8 + r) * COUT + c] = (_Float16)h;
      }
    }
  }
}

// ---------------------------------------------------------------------------
// EM stage 1: partial proto[b][k][c] = sum_n pm[b][k][n]*feats[b][n][c]
// over a 512-pixel chunk. Deterministic two-stage reduction (no atomics).
// grid = B*32 blocks, 256 threads; partials layout [B][32][5][64]
// ---------------------------------------------------------------------------
__global__ void __launch_bounds__(256) k_proto1(
    const float* __restrict__ pm, const _Float16* __restrict__ feats,
    float* __restrict__ partials) {
  const int b = blockIdx.x >> 5, ch = blockIdx.x & 31;
  const int t = threadIdx.x;
  const int c = t & 63, g = t >> 6;           // 4 pixel groups
  const int nbase = ch * 512;
  const float* pmb = pm + (long)b * KPROT * NPIX + nbase;
  const _Float16* fb = feats + ((long)b * NPIX + nbase) * COUT;
  float acc[KPROT] = {0.f, 0.f, 0.f, 0.f, 0.f};
  for (int p = g; p < 512; p += 4) {
    float f = (float)fb[(long)p * COUT + c];
#pragma unroll
    for (int k = 0; k < KPROT; ++k) acc[k] += pmb[(long)k * NPIX + p] * f;
  }
  __shared__ float sm[4][KPROT][64];
#pragma unroll
  for (int k = 0; k < KPROT; ++k) sm[g][k][c] = acc[k];
  __syncthreads();
  if (t < 320) {
    int k = t >> 6, cc = t & 63;
    float s = sm[0][k][cc] + sm[1][k][cc] + sm[2][k][cc] + sm[3][k][cc];
    partials[((long)b * 32 + ch) * 320 + k * 64 + cc] = s;
  }
}

// Reduce partials (mode 0) or take direct prototypes (mode 1), then
// normalize across K per (b,c): proto = v / (1e-8 + ||v||_K).
__global__ void __launch_bounds__(64) k_norm(
    const float* __restrict__ partials, const float* __restrict__ direct,
    float* __restrict__ proto, int mode) {
  const int b = blockIdx.x, c = threadIdx.x;
  float v[KPROT];
  if (mode == 0) {
#pragma unroll
    for (int k = 0; k < KPROT; ++k) v[k] = 0.f;
    for (int ch = 0; ch < 32; ++ch) {
      const float* p = partials + ((long)b * 32 + ch) * 320;
#pragma unroll
      for (int k = 0; k < KPROT; ++k) v[k] += p[k * 64 + c];
    }
  } else {
#pragma unroll
    for (int k = 0; k < KPROT; ++k) v[k] = direct[((long)b * KPROT + k) * 64 + c];
  }
  float n2 = 0.f;
#pragma unroll
  for (int k = 0; k < KPROT; ++k) n2 += v[k] * v[k];
  float inv = 1.0f / (1e-8f + sqrtf(n2));
#pragma unroll
  for (int k = 0; k < KPROT; ++k) proto[((long)b * KPROT + k) * 64 + c] = v[k] * inv;
}

// att = scale * proto @ feats^T ; pm = softmax_K(att) ; pm /= (1e-8 + sum_K pm)
// one thread per pixel; proto cached in LDS; feats reads are L2-resident f16.
__global__ void __launch_bounds__(256) k_att(
    const float* __restrict__ proto, const _Float16* __restrict__ feats,
    float* __restrict__ pmo) {
  __shared__ float pr[KPROT * 64];
  const long gp0 = (long)blockIdx.x * 256;
  const int b = (int)(gp0 >> 14);
  const int t = threadIdx.x;
  for (int i = t; i < KPROT * 64; i += 256) pr[i] = proto[(long)b * KPROT * 64 + i];
  __syncthreads();
  const long gpix = gp0 + t;
  const int n = (int)(gpix & (NPIX - 1));
  const _Float16* f = feats + gpix * COUT;
  float a[KPROT] = {0.f, 0.f, 0.f, 0.f, 0.f};
#pragma unroll
  for (int cc = 0; cc < 8; ++cc) {
    v8h fv = *(const v8h*)(f + cc * 8);
#pragma unroll
    for (int e = 0; e < 8; ++e) {
      float xv = (float)fv[e];
      int c = cc * 8 + e;
#pragma unroll
      for (int k = 0; k < KPROT; ++k) a[k] += pr[k * 64 + c] * xv;
    }
  }
  const float scale = 0.125f;  // 64^-0.5
  float mx = -3.4e38f;
#pragma unroll
  for (int k = 0; k < KPROT; ++k) { a[k] *= scale; mx = a[k] > mx ? a[k] : mx; }
  float e[KPROT], se = 0.f;
#pragma unroll
  for (int k = 0; k < KPROT; ++k) { e[k] = __expf(a[k] - mx); se += e[k]; }
  float inv = 1.0f / se, s2 = 0.f;
#pragma unroll
  for (int k = 0; k < KPROT; ++k) { e[k] *= inv; s2 += e[k]; }
  float inv2 = 1.0f / (1e-8f + s2);
#pragma unroll
  for (int k = 0; k < KPROT; ++k)
    pmo[((long)b * KPROT + k) * NPIX + n] = e[k] * inv2;
}

// context[b][n][colofs + c] = sum_k proto[b][k][c] * pm[b][k][n]  (f16 out)
__global__ void __launch_bounds__(256) k_ctx(
    const float* __restrict__ proto, const float* __restrict__ pm,
    _Float16* __restrict__ ctx, int colofs) {
  __shared__ float pr[KPROT * 64];
  const long gp0 = (long)blockIdx.x * 256;
  const int b = (int)(gp0 >> 14);
  const int t = threadIdx.x;
  for (int i = t; i < KPROT * 64; i += 256) pr[i] = proto[(long)b * KPROT * 64 + i];
  __syncthreads();
  const long gpix = gp0 + t;
  const int n = (int)(gpix & (NPIX - 1));
  float w[KPROT];
#pragma unroll
  for (int k = 0; k < KPROT; ++k) w[k] = pm[((long)b * KPROT + k) * NPIX + n];
  _Float16* o = ctx + gpix * 128 + colofs;
#pragma unroll
  for (int cc = 0; cc < 8; ++cc) {
    v8h o8;
#pragma unroll
    for (int e = 0; e < 8; ++e) {
      int c = cc * 8 + e;
      float s = 0.f;
#pragma unroll
      for (int k = 0; k < KPROT; ++k) s += pr[k * 64 + c] * w[k];
      o8[e] = (_Float16)s;
    }
    *(v8h*)(o + cc * 8) = o8;
  }
}

// ---------------------------------------------------------------------------
// Final conv: out = relu(BN(ctx[B*N,128] @ W2^T[128,64])), output [B][64][N]
// WMMA with A fragments loaded straight from L2-resident ctx, output
// transposed through LDS for coalesced [C][N] stores.
// ---------------------------------------------------------------------------
__global__ void __launch_bounds__(256) k_final(
    const _Float16* __restrict__ ctx, const _Float16* __restrict__ W2,
    const float* __restrict__ so, float* __restrict__ out) {
  __shared__ __align__(16) float os[64 * 132];
  const int t = threadIdx.x;
  const int wave = t >> 5, lane = t & 31;
  const int ln = lane & 15, hf = lane >> 4;
  const long pixbase = (long)blockIdx.x * 128;
  const _Float16* ab = ctx + pixbase * 128;
  v8f acc[4];
#pragma unroll
  for (int j = 0; j < 4; ++j) acc[j] = zero8();
#pragma unroll
  for (int kc = 0; kc < 4; ++kc) {
    const _Float16* ap = ab + (long)(wave * 16 + ln) * 128 + kc * 32 + hf * 8;
    v16h a = frag_split(ap, ap + 16);
#pragma unroll
    for (int j = 0; j < 4; ++j) {
      v16h bb = frag_split(W2 + (long)(j * 16 + ln) * 128 + kc * 32 + hf * 8, W2 + (long)(j * 16 + ln) * 128 + kc * 32 + hf * 8 + 16);
      acc[j] = wmma_f16(a, bb, acc[j]);
    }
  }
#pragma unroll
  for (int j = 0; j < 4; ++j) {
    int c = j * 16 + ln;
    float s = so[512 + c], o = so[512 + 64 + c];
#pragma unroll
    for (int r = 0; r < 8; ++r) {
      float h = acc[j][r] * s + o;
      h = h > 0.f ? h : 0.f;
      os[c * 132 + wave * 16 + hf * 8 + r] = h;
    }
  }
  __syncthreads();
  const int b = (int)(pixbase >> 14);
  const int n0 = (int)(pixbase & (NPIX - 1));
  float* ob = out + (long)b * COUT * NPIX + n0;
  const int c = t >> 2, q = t & 3;
#pragma unroll
  for (int i = 0; i < 8; ++i) {
    v4f v = *(const v4f*)&os[c * 132 + q * 32 + i * 4];
    *(v4f*)&ob[(long)c * NPIX + q * 32 + i * 4] = v;
  }
}

// ---------------------------------------------------------------------------
// Workspace layout (bytes)
// ---------------------------------------------------------------------------
static constexpr size_t OFF_WH    = 0;                        // 49152 halfs
static constexpr size_t OFF_SO    = 98304;                    // 640 f32
static constexpr size_t OFF_F0    = 102400;                   // 33554432 B
static constexpr size_t OFF_F1    = OFF_F0   + 33554432;
static constexpr size_t OFF_CTX   = OFF_F1   + 33554432;      // 67108864 B
static constexpr size_t OFF_PML   = OFF_CTX  + 67108864;      // 5242880 B
static constexpr size_t OFF_PMG   = OFF_PML  + 5242880;
static constexpr size_t OFF_PROTO = OFF_PMG  + 5242880;       // 20480 B
static constexpr size_t OFF_GP    = OFF_PROTO + 20480;
static constexpr size_t OFF_PART  = OFF_GP   + 20480;         // 655360 B

extern "C" void kernel_launch(void* const* d_in, const int* in_sizes, int n_in,
                              void* d_out, int out_size, void* d_ws, size_t ws_size,
                              hipStream_t stream) {
  (void)in_sizes; (void)n_in; (void)out_size; (void)ws_size;
  const float* x     = (const float*)d_in[0];
  const float* cmask = (const float*)d_in[1];
  const float* gprot = (const float*)d_in[2];
  const float* W0a = (const float*)d_in[3],  *b0a = (const float*)d_in[4],  *bn0a = (const float*)d_in[5];
  const float* W0b = (const float*)d_in[6],  *b0b = (const float*)d_in[7],  *bn0b = (const float*)d_in[8];
  const float* W1a = (const float*)d_in[9],  *b1a = (const float*)d_in[10], *bn1a = (const float*)d_in[11];
  const float* W1b = (const float*)d_in[12], *b1b = (const float*)d_in[13], *bn1b = (const float*)d_in[14];
  const float* W2  = (const float*)d_in[15], *b2  = (const float*)d_in[16], *bn2  = (const float*)d_in[17];
  float* out = (float*)d_out;

  char* ws = (char*)d_ws;
  _Float16* wh   = (_Float16*)(ws + OFF_WH);
  float*    so   = (float*)   (ws + OFF_SO);
  _Float16* f0   = (_Float16*)(ws + OFF_F0);
  _Float16* f1   = (_Float16*)(ws + OFF_F1);
  _Float16* ctx  = (_Float16*)(ws + OFF_CTX);
  float*    pml  = (float*)   (ws + OFF_PML);
  float*    pmg  = (float*)   (ws + OFF_PMG);
  float*    prot = (float*)   (ws + OFF_PROTO);
  float*    gp   = (float*)   (ws + OFF_GP);
  float*    part = (float*)   (ws + OFF_PART);

  k_setup<<<1, 256, 0, stream>>>(W0a, b0a, bn0a, W0b, b0b, bn0b,
                                 W1a, b1a, bn1a, W1b, b1b, bn1b,
                                 W2,  b2,  bn2,  wh, so);

  // fused dual stem (reads x once)
  k_stem<<<2048, 256, 0, stream>>>(x, wh, wh + 16384, wh + 20480, wh + 36864,
                                   so, f0, f1);

  // --- local branch EM (pm initialized from coarse_mask) ---
  const float* pmin = cmask;
  for (int it = 0; it < 3; ++it) {
    k_proto1<<<BATCH * 32, 256, 0, stream>>>(pmin, f0, part);
    k_norm<<<BATCH, 64, 0, stream>>>(part, nullptr, prot, 0);
    k_att<<<BATCH * NPIX / 256, 256, 0, stream>>>(prot, f0, pml);
    pmin = pml;
  }
  k_ctx<<<BATCH * NPIX / 256, 256, 0, stream>>>(prot, pml, ctx, 0);

  // --- global branch EM (seeded by normalized global prototypes) ---
  k_norm<<<BATCH, 64, 0, stream>>>(nullptr, gprot, gp, 1);
  for (int it = 0; it < 3; ++it) {
    k_att<<<BATCH * NPIX / 256, 256, 0, stream>>>(gp, f1, pmg);
    k_proto1<<<BATCH * 32, 256, 0, stream>>>(pmg, f1, part);
    k_norm<<<BATCH, 64, 0, stream>>>(part, nullptr, gp, 0);
  }
  k_ctx<<<BATCH * NPIX / 256, 256, 0, stream>>>(gp, pmg, ctx, 64);

  // final 1x1 conv + BN + ReLU
  k_final<<<2048, 256, 0, stream>>>(ctx, wh + 40960, so, out);
}
